// GDBlock_38508676776441
// MI455X (gfx1250) — compile-verified, matrix operand order checked by reading, not run on hardware
//
#include <hip/hip_runtime.h>
#include <hip/hip_bf16.h>
#include <math.h>

// ---------------- problem constants ----------------
constexpr int B_ = 2, S_ = 2048, E_ = 1024, H_ = 16, DFF_ = 4096, V_ = 32000;

// ---------------- WMMA types ----------------
typedef __attribute__((ext_vector_type(16))) __bf16 v16bf;
typedef __attribute__((ext_vector_type(8)))  float  v8f;

constexpr int TBM = 128, TBN = 128, TBK = 32, TBKP = 40; // padded LDS k-stride (bf16 elems)

union AccU { v8f v; float f[8]; };

// Load one 16x32 bf16 fragment (A or B role) from a row-major LDS tile [rows][TBKP].
__device__ inline v16bf load_frag(const __hip_bfloat16* lds, int fragrow, int lane) {
    int row = fragrow + (lane & 15);
    int k0  = (lane >> 4) << 3;                // 0 or 8
    const __hip_bfloat16* p = lds + row * TBKP + k0;
    union { v16bf v; uint4 u[2]; } f;
    f.u[0] = *(const uint4*)(p);
    f.u[1] = *(const uint4*)(p + 16);
    return f.v;
}

// Epilogue IDs
// 0: plain f32 store
// 1: t = exp(clip(x,-10,10)); store bf16; atomic rowsum into aux0
// 2: x / (aux0[m] + 1e-8); f32 store
// 3: x * aux0[z & zmask]; bf16 store       (A_lr fold for Vh)
// 4: aux0[m*ldc+n] + (x + aux1[z*aux1_z+n]) * scale; f32 store   (f_k + (dA+dB)/S)
// 5: exact GELU; bf16 store
template<bool ABF, bool BBF, bool BNT, int EPI>
__global__ __launch_bounds__(256) void gemm_wmma_k(
    const void* __restrict__ Ag, const void* __restrict__ Bg, void* __restrict__ Cg,
    int K,
    long long a_rs, int a_kshift, long long a_bs, long long a_z, int a_zshift,
    long long b_rs, long long b_z, int b_zmask,
    long long c_ld, long long c_z,
    float* aux0, const float* aux1, long long aux1_z,
    float scale, int zmask)
{
    __shared__ __align__(16) __hip_bfloat16 As[2][TBM * TBKP];
    __shared__ __align__(16) __hip_bfloat16 Bs[2][TBN * TBKP];

    const int tid  = threadIdx.x;
    const int z    = blockIdx.z;
    const int lane = tid & 31;
    const int wave = tid >> 5;
    const int wm   = (wave & 3) * 32;   // 4 wave-rows  x 32 rows
    const int wn   = (wave >> 2) * 64;  // 2 wave-cols  x 64 cols

    const long long a_kmask = (1LL << a_kshift) - 1;
    const long long abase = (long long)blockIdx.y * TBM * a_rs + (long long)(z >> a_zshift) * a_z;
    const long long bbase = BNT
        ? (long long)blockIdx.x * TBN * b_rs + (long long)(z & b_zmask) * b_z
        : (long long)blockIdx.x * TBN        + (long long)(z & b_zmask) * b_z;

    // ---- staging helpers ----
    auto stageA = [&](__hip_bfloat16* dst, int kb) {
        int r  = tid >> 1;
        int kc = (tid & 1) << 4;            // 0 or 16
        long long kg = (long long)kb + kc;
        long long o  = abase + (long long)r * a_rs + ((kg >> a_kshift) * a_bs) + (kg & a_kmask);
        __hip_bfloat16* d = dst + r * TBKP + kc;
        if (ABF) {
            const __hip_bfloat16* g = (const __hip_bfloat16*)Ag + o;
            unsigned l = (unsigned)(size_t)(void*)d;   // flat->LDS offset (addr[31:0])
            asm volatile("global_load_async_to_lds_b128 %0, %1, off"
                         :: "v"(l), "v"(g) : "memory");
            asm volatile("global_load_async_to_lds_b128 %0, %1, off"
                         :: "v"(l + 16u), "v"(g + 8) : "memory");
            __builtin_prefetch((const char*)(g + TBK), 0, 1);
        } else {
            const float4* s = (const float4*)((const float*)Ag + o);
#pragma unroll
            for (int i = 0; i < 4; ++i) {
                float4 fv = s[i];
                d[i * 4 + 0] = __float2bfloat16(fv.x);
                d[i * 4 + 1] = __float2bfloat16(fv.y);
                d[i * 4 + 2] = __float2bfloat16(fv.z);
                d[i * 4 + 3] = __float2bfloat16(fv.w);
            }
            __builtin_prefetch((const char*)(s + 8), 0, 1);   // + TBK floats
        }
    };
    auto stageB = [&](__hip_bfloat16* dst, int kb) {
        if (BNT) {  // source [N,K] row-major (K contiguous)
            int r  = tid >> 1;
            int kc = (tid & 1) << 4;
            long long o = bbase + (long long)r * b_rs + kb + kc;
            __hip_bfloat16* d = dst + r * TBKP + kc;
            if (BBF) {
                const uint4* s = (const uint4*)((const __hip_bfloat16*)Bg + o);
                uint4 x0 = s[0], x1 = s[1];
                *(uint4*)(d) = x0; *(uint4*)(d + 8) = x1;
                __builtin_prefetch((const char*)(s + 4), 0, 1);
            } else {
                const float4* s = (const float4*)((const float*)Bg + o);
#pragma unroll
                for (int i = 0; i < 4; ++i) {
                    float4 fv = s[i];
                    d[i * 4 + 0] = __float2bfloat16(fv.x);
                    d[i * 4 + 1] = __float2bfloat16(fv.y);
                    d[i * 4 + 2] = __float2bfloat16(fv.z);
                    d[i * 4 + 3] = __float2bfloat16(fv.w);
                }
                __builtin_prefetch((const char*)(s + 8), 0, 1);
            }
        } else {    // source [K,N] row-major: transpose while staging
            int n  = tid & 127;
            int k0 = (tid >> 7) << 4;       // 0 or 16
#pragma unroll
            for (int i = 0; i < 16; ++i) {
                long long o = bbase + (long long)(kb + k0 + i) * b_rs + n;
                if (BBF) dst[n * TBKP + k0 + i] = ((const __hip_bfloat16*)Bg)[o];
                else     dst[n * TBKP + k0 + i] = __float2bfloat16(((const float*)Bg)[o]);
            }
        }
    };

    v8f zero = {};
    v8f acc[2][4];
#pragma unroll
    for (int i = 0; i < 2; ++i)
#pragma unroll
        for (int j = 0; j < 4; ++j) acc[i][j] = zero;

    const int nk = K / TBK;

    // prologue: fill buffer 0
    stageA(&As[0][0], 0);
    stageB(&Bs[0][0], 0);
    if (ABF) asm volatile("s_wait_asynccnt 0x0" ::: "memory");
    __syncthreads();

    for (int it = 0; it < nk; ++it) {
        const int cur = it & 1;
        const int nxt = cur ^ 1;
        if (it + 1 < nk) {                  // stage next tile into the other buffer
            stageA(&As[nxt][0], (it + 1) * TBK);
            stageB(&Bs[nxt][0], (it + 1) * TBK);
        }

        const __hip_bfloat16* Ac = &As[cur][0];
        const __hip_bfloat16* Bc = &Bs[cur][0];
        v16bf a0 = load_frag(Ac, wm,      lane);
        v16bf a1 = load_frag(Ac, wm + 16, lane);
        v16bf b0 = load_frag(Bc, wn,      lane);
        v16bf b1 = load_frag(Bc, wn + 16, lane);
        v16bf b2 = load_frag(Bc, wn + 32, lane);
        v16bf b3 = load_frag(Bc, wn + 48, lane);

        acc[0][0] = __builtin_amdgcn_wmma_f32_16x16x32_bf16(false, a0, false, b0, (short)0, acc[0][0], false, false);
        acc[0][1] = __builtin_amdgcn_wmma_f32_16x16x32_bf16(false, a0, false, b1, (short)0, acc[0][1], false, false);
        acc[0][2] = __builtin_amdgcn_wmma_f32_16x16x32_bf16(false, a0, false, b2, (short)0, acc[0][2], false, false);
        acc[0][3] = __builtin_amdgcn_wmma_f32_16x16x32_bf16(false, a0, false, b3, (short)0, acc[0][3], false, false);
        acc[1][0] = __builtin_amdgcn_wmma_f32_16x16x32_bf16(false, a1, false, b0, (short)0, acc[1][0], false, false);
        acc[1][1] = __builtin_amdgcn_wmma_f32_16x16x32_bf16(false, a1, false, b1, (short)0, acc[1][1], false, false);
        acc[1][2] = __builtin_amdgcn_wmma_f32_16x16x32_bf16(false, a1, false, b2, (short)0, acc[1][2], false, false);
        acc[1][3] = __builtin_amdgcn_wmma_f32_16x16x32_bf16(false, a1, false, b3, (short)0, acc[1][3], false, false);

        if (ABF) asm volatile("s_wait_asynccnt 0x0" ::: "memory");
        __syncthreads();                    // publish buffer 'nxt', retire reads of 'cur'
    }

    // ---- epilogue ----
    const long long m0 = (long long)blockIdx.y * TBM + wm + ((lane >> 4) << 3);
    const long long n0 = (long long)blockIdx.x * TBN + wn + (lane & 15);
    const long long cz = (long long)z * c_z;

#pragma unroll
    for (int mi = 0; mi < 2; ++mi) {
        float rp[8];
#pragma unroll
        for (int v = 0; v < 8; ++v) rp[v] = 0.f;
#pragma unroll
        for (int ni = 0; ni < 4; ++ni) {
            AccU u; u.v = acc[mi][ni];
#pragma unroll
            for (int v = 0; v < 8; ++v) {
                long long m = m0 + mi * 16 + v;
                long long n = n0 + ni * 16;
                float x = u.f[v];
                if (EPI == 0) {
                    ((float*)Cg)[cz + m * c_ld + n] = x;
                } else if (EPI == 1) {
                    float t = __expf(fminf(fmaxf(x, -10.f), 10.f));
                    ((__hip_bfloat16*)Cg)[cz + m * c_ld + n] = __float2bfloat16(t);
                    rp[v] += t;
                } else if (EPI == 2) {
                    ((float*)Cg)[cz + m * c_ld + n] = x / (aux0[m] + 1e-8f);
                } else if (EPI == 3) {
                    ((__hip_bfloat16*)Cg)[cz + m * c_ld + n] = __float2bfloat16(x * aux0[z & zmask]);
                } else if (EPI == 4) {
                    float o = aux0[m * c_ld + n] + (x + aux1[(long long)z * aux1_z + n]) * scale;
                    ((float*)Cg)[cz + m * c_ld + n] = o;
                } else if (EPI == 5) {
                    float g = 0.5f * x * (1.0f + erff(x * 0.70710678118f));
                    ((__hip_bfloat16*)Cg)[cz + m * c_ld + n] = __float2bfloat16(g);
                }
            }
        }
        if (EPI == 1) {
#pragma unroll
            for (int v = 0; v < 8; ++v) {
                float r = rp[v];
                r += __shfl_xor(r, 1, 32);
                r += __shfl_xor(r, 2, 32);
                r += __shfl_xor(r, 4, 32);
                r += __shfl_xor(r, 8, 32);
                if ((lane & 15) == 0) atomicAdd(aux0 + (m0 + mi * 16 + v), r);
            }
        }
    }
}

// diff = e - E_W_e (broadcast over batch); deltaB[b,e] += B_lr * partial sum_s diff
__global__ __launch_bounds__(256) void diff_kernel(
    const float* __restrict__ e_in, const float* __restrict__ ewe,
    const float* __restrict__ B_lr, float* __restrict__ diff, float* __restrict__ deltaB)
{
    int ei = blockIdx.x * 256 + threadIdx.x;
    int b  = blockIdx.y;
    int s0 = blockIdx.z * (S_ / 16);
    float accv = 0.f;
    for (int s = s0; s < s0 + S_ / 16; ++s) {
        float d = e_in[((long long)b * S_ + s) * E_ + ei] - ewe[(long long)s * E_ + ei];
        diff[((long long)b * S_ + s) * E_ + ei] = d;
        accv += d;
    }
    atomicAdd(&deltaB[b * E_ + ei], accv * B_lr[0]);
}

extern "C" void kernel_launch(void* const* d_in, const int* in_sizes, int n_in,
                              void* d_out, int out_size, void* d_ws, size_t ws_size,
                              hipStream_t stream)
{
    const float* f_k   = (const float*)d_in[0];
    const float* attn  = (const float*)d_in[1];
    const float* e_in  = (const float*)d_in[2];
    const float* W_v   = (const float*)d_in[3];
    const float* W_e   = (const float*)d_in[4];
    const float* A_lr  = (const float*)d_in[5];
    const float* B_lr  = (const float*)d_in[6];
    const float* ff_w1 = (const float*)d_in[7];
    const float* ff_w2 = (const float*)d_in[8];

    char* ws = (char*)d_ws;
    size_t off = 0;
    auto alloc = [&](size_t bytes) -> void* {
        void* p = ws + off;
        off += (bytes + 255) & ~(size_t)255;
        return p;
    };
    __hip_bfloat16* T      = (__hip_bfloat16*)alloc((size_t)S_ * V_ * 2);           // 131 MB
    float*          rowsum = (float*)alloc((size_t)S_ * 4);                          // 8 KB
    float*          deltaB = (float*)alloc((size_t)B_ * E_ * 4);                     // 8 KB (adjacent: one memset)
    float*          EWE    = (float*)alloc((size_t)S_ * E_ * 4);                     // 8 MB
    float*          diff   = (float*)alloc((size_t)B_ * S_ * E_ * 4);                // 16 MB
    __hip_bfloat16* Vh     = (__hip_bfloat16*)alloc((size_t)B_ * H_ * S_ * E_ * 2);  // 134 MB
    float*          outp   = (float*)alloc((size_t)B_ * S_ * E_ * 4);                // 16 MB
    __hip_bfloat16* hbuf   = (__hip_bfloat16*)alloc((size_t)B_ * S_ * DFF_ * 2);     // 33 MB

    dim3 blk(256);

    hipMemsetAsync(rowsum, 0, (size_t)(S_ + B_ * E_) * 4, stream);  // rowsum + deltaB

    // 1) T = exp(clip(f_k @ W_e^T)), rowsum  : M=S, N=V, K=E  (NT)
    gemm_wmma_k<false, false, true, 1><<<dim3(V_ / TBN, S_ / TBM, 1), blk, 0, stream>>>(
        f_k, W_e, T, E_,
        E_, 40, 0, 0, 0,
        E_, 0, 0,
        V_, 0,
        rowsum, nullptr, 0, 0.f, 0);

    // 2) E_W_e = (T @ W_e) / rowsum : M=S, N=E, K=V  (NN, A bf16 via async-LDS)
    gemm_wmma_k<true, false, false, 2><<<dim3(E_ / TBN, S_ / TBM, 1), blk, 0, stream>>>(
        T, W_e, EWE, V_,
        V_, 40, 0, 0, 0,
        E_, 0, 0,
        E_, 0,
        rowsum, nullptr, 0, 0.f, 0);

    // 3) diff + deltaB
    diff_kernel<<<dim3(E_ / 256, B_, 16), blk, 0, stream>>>(e_in, EWE, B_lr, diff, deltaB);

    // 4) Vh_scaled[b,h] = (diff[b] @ W_v[h]) * A_lr[h] : z = b*H+h, M=S, N=E, K=E (NN)
    gemm_wmma_k<false, false, false, 3><<<dim3(E_ / TBN, S_ / TBM, B_ * H_), blk, 0, stream>>>(
        diff, W_v, Vh, E_,
        E_, 40, 0, (long long)S_ * E_, 4,
        E_, (long long)E_ * E_, H_ - 1,
        E_, (long long)S_ * E_,
        (float*)A_lr, nullptr, 0, 0.f, H_ - 1);

    // 5) out_pre[b] = f_k + (attn[b](S x H*S) @ Vh[b](H*S x E) + deltaB[b]) / S
    //    z = b, M=S, N=E, K=H*S; k>>11 folds the h sum into K. (NN, B bf16)
    gemm_wmma_k<false, true, false, 4><<<dim3(E_ / TBN, S_ / TBM, B_), blk, 0, stream>>>(
        attn, Vh, outp, H_ * S_,
        S_, 11, (long long)S_ * S_, (long long)H_ * S_ * S_, 0,
        E_, (long long)H_ * S_ * E_, -1,
        E_, (long long)S_ * E_,
        (float*)f_k, deltaB, E_, 1.0f / (float)S_, 0);

    // 6) h = gelu(out_pre @ ff_w1^T) : M=B*S, N=DFF, K=E (NT)
    gemm_wmma_k<false, false, true, 5><<<dim3(DFF_ / TBN, (B_ * S_) / TBM, 1), blk, 0, stream>>>(
        outp, ff_w1, hbuf, E_,
        E_, 40, 0, 0, 0,
        E_, 0, 0,
        DFF_, 0,
        nullptr, nullptr, 0, 0.f, 0);

    // 7) out = h @ ff_w2^T : M=B*S, N=E, K=DFF (NT, A bf16 via async-LDS) -> d_out f32
    gemm_wmma_k<true, false, true, 0><<<dim3(E_ / TBN, (B_ * S_) / TBM, 1), blk, 0, stream>>>(
        hbuf, ff_w2, d_out, DFF_,
        DFF_, 40, 0, 0, 0,
        DFF_, 0, 0,
        E_, 0,
        nullptr, nullptr, 0, 0.f, 1);
}
